// MCAT_23630910062939
// MI455X (gfx1250) — compile-verified, matrix operand order checked by reading, not run on hardware
//
#include <hip/hip_runtime.h>
#include <hip/hip_bf16.h>

// ---------------------------------------------------------------------------
// Shared-weight transformer encoder block (x2) for MI455X / gfx1250 (wave32).
// All GEMM-shaped math runs on v_wmma_f32_16x16x32_f16. WMMA operands staged
// in LDS in fragment-friendly layouts (2x ds_load_b128 per fragment).
// Loop top does clause'd global_load_b128 -> ds_store_b128 (no register
// pipeline: the RA spills long-lived staging regs to scratch, so we rely on
// global_prefetch_b8 two K-tiles ahead + multi-wave occupancy instead).
// WMMA bursts are split into two 4-frag groups + sched_barrier so dscnt
// waits stagger instead of serializing each WMMA.
// ---------------------------------------------------------------------------

typedef __attribute__((ext_vector_type(16))) _Float16 v16h;
typedef __attribute__((ext_vector_type(8)))  float    v8f;
typedef _Float16 half_t;

static constexpr int B  = 8;
static constexpr int S  = 1024;
static constexpr int D  = 1024;
static constexpr int H  = 16;
static constexpr int DK = 64;
static constexpr int SP = S + 4;          // padded seq (2 each side, covers f=5)
static constexpr int M_TOT = B * S;       // 8192
static constexpr int LDT = 72;            // 64 + 8 halves pad (16B, keeps b128 align)
static constexpr int LDK = 72;            // K-chunk tile pad
static constexpr int LDV = 40;            // 32 + 8 halves pad

#define WMMA16(a, b, c) \
  __builtin_amdgcn_wmma_f32_16x16x32_f16(false, (a), false, (b), (short)0, (c), false, false)

#if __has_builtin(__builtin_amdgcn_sched_barrier)
#define SCHED_FENCE() __builtin_amdgcn_sched_barrier(0)
#else
#define SCHED_FENCE()
#endif

// Fragment gather from a row-major tile where each lane's 16 halves are two
// contiguous 8-half chunks (lowers to 2x ds_load_b128).
__device__ __forceinline__ v16h frag_rm(const half_t* p, int ld, int lane) {
  const int row = lane & 15;
  const int kb  = (lane >> 4) << 3;
  const half_t* q = p + row * ld + kb;
  v16h a;
#pragma unroll
  for (int j = 0; j < 8; ++j) { a[j] = q[j]; a[j + 8] = q[j + 16]; }
  return a;
}

// ---------------------------------------------------------------------------
// LayerNorm (torch variant: unbiased std, eps added to std).
// mode 0: f16 flat [B*S, D]   mode 1: f16 into padded [B, S+4, D] (+2 rows)
// mode 2: f32 flat (final output)
// ---------------------------------------------------------------------------
__global__ __launch_bounds__(256) void k_ln(const float* __restrict__ x,
                                            const float* __restrict__ ga,
                                            const float* __restrict__ gb,
                                            half_t* __restrict__ o16,
                                            float* __restrict__ o32, int mode) {
  const int row = blockIdx.x;                 // 0 .. B*S-1
  const float* xr = x + (size_t)row * D;
  __shared__ float red[2][8];
  float s = 0.f, sq = 0.f;
  for (int i = threadIdx.x; i < D; i += 256) { float v = xr[i]; s += v; sq += v * v; }
#pragma unroll
  for (int off = 16; off > 0; off >>= 1) { s += __shfl_down(s, off); sq += __shfl_down(sq, off); }
  const int wid = threadIdx.x >> 5, lane = threadIdx.x & 31;
  if (lane == 0) { red[0][wid] = s; red[1][wid] = sq; }
  __syncthreads();
  if (threadIdx.x == 0) {
    float ts = 0.f, tq = 0.f;
    for (int w = 0; w < 8; ++w) { ts += red[0][w]; tq += red[1][w]; }
    red[0][0] = ts; red[1][0] = tq;
  }
  __syncthreads();
  s = red[0][0]; sq = red[1][0];
  const float mean = s / (float)D;
  const float var  = (sq - s * mean) / (float)(D - 1);         // unbiased
  const float inv  = 1.0f / (sqrtf(fmaxf(var, 0.f)) + 1e-6f);  // eps on std
  size_t dbase;
  if (mode == 1) {
    const int b = row >> 10, sr = row & (S - 1);
    dbase = ((size_t)b * SP + sr + 2) * D;
  } else {
    dbase = (size_t)row * D;
  }
  for (int i = threadIdx.x; i < D; i += 256) {
    const float y = ga[i] * (xr[i] - mean) * inv + gb[i];
    if (mode == 2) o32[dbase + i] = y;
    else           o16[dbase + i] = (half_t)y;
  }
}

// ---------------------------------------------------------------------------
// GEMM: C[M x 1024] = A[M x 1024](f16) * Wt^T (Wt[n][k]) + bias
// Block = 128 threads (4 waves), tile 64(M) x 64(N), K-step 64.
// mode 0: out16 = f16(acc + bias)   mode 1: acc32 += acc + bias (residual)
// ---------------------------------------------------------------------------
__global__ __launch_bounds__(128)
void k_gemm(const half_t* __restrict__ A,
            const half_t* __restrict__ Wt,
            const float* __restrict__ bias,
            half_t* __restrict__ out16,
            float* __restrict__ acc32, int mode) {
  __shared__ __align__(16) half_t lA[64 * LDT];
  __shared__ __align__(16) half_t lB[64 * LDT];
  const int n0 = blockIdx.x * 64, m0 = blockIdx.y * 64;
  const int tid = threadIdx.x, wid = tid >> 5, lane = tid & 31;
  const int lr = (tid * 32) >> 6, lc = (tid * 32) & 63;      // lc in {0,32}
  const half_t* gA = A  + (size_t)(m0 + lr) * D + lc;
  const half_t* gW = Wt + (size_t)(n0 + lr) * D + lc;
  uint4* sa = (uint4*)(lA + lr * LDT + lc);
  uint4* sb = (uint4*)(lB + lr * LDT + lc);

  // warm the cache two tiles deep before the loop
  __builtin_prefetch(gA, 0, 1);
  __builtin_prefetch(gW, 0, 1);
  __builtin_prefetch(gA + 64, 0, 1);
  __builtin_prefetch(gW + 64, 0, 1);

  v8f acc[4] = {};
  for (int k0 = 0; k0 < D; k0 += 64) {
    __syncthreads();                                         // prev tile fully read
    {                                                        // clause'd b128 loads -> LDS
      const uint4* p = (const uint4*)(gA + k0);
      sa[0] = p[0]; sa[1] = p[1]; sa[2] = p[2]; sa[3] = p[3];
      const uint4* q = (const uint4*)(gW + k0);
      sb[0] = q[0]; sb[1] = q[1]; sb[2] = q[2]; sb[3] = q[3];
    }
    __syncthreads();                                         // tile visible
    if (k0 + 128 < D) {                                      // prefetch 2 tiles ahead
      __builtin_prefetch(gA + k0 + 128, 0, 1);
      __builtin_prefetch(gW + k0 + 128, 0, 1);
    }
    // group 1: a0 x 4 N-tiles (low K half)
    {
      const v16h a0 = frag_rm(lA + wid * 16 * LDT, LDT, lane);
      v16h bf[4];
#pragma unroll
      for (int ns = 0; ns < 4; ++ns) bf[ns] = frag_rm(lB + ns * 16 * LDT, LDT, lane);
      SCHED_FENCE();
#pragma unroll
      for (int ns = 0; ns < 4; ++ns) acc[ns] = WMMA16(a0, bf[ns], acc[ns]);
    }
    // group 2: a1 x 4 N-tiles (high K half); loads overlap group-1 WMMAs
    {
      const v16h a1 = frag_rm(lA + wid * 16 * LDT + 32, LDT, lane);
      v16h bf[4];
#pragma unroll
      for (int ns = 0; ns < 4; ++ns) bf[ns] = frag_rm(lB + ns * 16 * LDT + 32, LDT, lane);
      SCHED_FENCE();
#pragma unroll
      for (int ns = 0; ns < 4; ++ns) acc[ns] = WMMA16(a1, bf[ns], acc[ns]);
    }
  }
  const int hi = lane >> 4, nl = lane & 15;
#pragma unroll
  for (int ns = 0; ns < 4; ++ns) {
    const int n = n0 + ns * 16 + nl;
    const float bv = bias[n];
#pragma unroll
    for (int r = 0; r < 8; ++r) {
      const size_t m = (size_t)(m0 + wid * 16 + r + 8 * hi);
      const float v = acc[ns][r] + bv;
      if (mode == 0) out16[m * D + n] = (half_t)v;
      else           acc32[m * D + n] += v;
    }
  }
}

// ---------------------------------------------------------------------------
// Conv branch as tap-shifted GEMM over padded activations; BN pre-folded.
// Wf layout: [f][n][k] (transposed per tap). Tap x K loop flattened.
// out = (first?0:out) + relu(conv + bias)/3
// ---------------------------------------------------------------------------
__global__ __launch_bounds__(128)
void k_conv(const half_t* __restrict__ Xp, // [B][SP][D]
            const half_t* __restrict__ Wf, // [f][D][D] (n,k)
            const float* __restrict__ bias,
            float* __restrict__ out, int f, int first) {
  __shared__ __align__(16) half_t lA[64 * LDT];
  __shared__ __align__(16) half_t lB[64 * LDT];
  const int n0 = blockIdx.x * 64, m0 = blockIdx.y * 64;
  const int b = m0 >> 10, sr0 = m0 & (S - 1);
  const int c = f >> 1;
  const int tid = threadIdx.x, wid = tid >> 5, lane = tid & 31;
  const int lr = (tid * 32) >> 6, lc = (tid * 32) & 63;
  uint4* sa = (uint4*)(lA + lr * LDT + lc);
  uint4* sb = (uint4*)(lB + lr * LDT + lc);
  const half_t* baseA = Xp + ((size_t)b * SP + sr0 + 2 - c + lr) * D + lc; // + t*D + k0
  const half_t* baseW = Wf + (size_t)(n0 + lr) * D + lc;                   // + t*D*D + k0

  __builtin_prefetch(baseA, 0, 1);
  __builtin_prefetch(baseW, 0, 1);

  const int NST = f * (D / 64);           // flattened steps: st -> (t = st/16, k0 = (st%16)*64)
  v8f acc[4] = {};
  for (int st = 0; st < NST; ++st) {
    const int t = st >> 4, k0 = (st & 15) << 6;
    __syncthreads();
    {
      const uint4* p = (const uint4*)(baseA + (size_t)t * D + k0);
      sa[0] = p[0]; sa[1] = p[1]; sa[2] = p[2]; sa[3] = p[3];
      const uint4* q = (const uint4*)(baseW + (size_t)t * D * D + k0);
      sb[0] = q[0]; sb[1] = q[1]; sb[2] = q[2]; sb[3] = q[3];
    }
    __syncthreads();
    if (st + 1 < NST) {                 // prefetch next step's lines
      const int t1 = (st + 1) >> 4, k1 = ((st + 1) & 15) << 6;
      __builtin_prefetch(baseA + (size_t)t1 * D + k1, 0, 1);
      __builtin_prefetch(baseW + (size_t)t1 * D * D + k1, 0, 1);
    }
    {
      const v16h a0 = frag_rm(lA + wid * 16 * LDT, LDT, lane);
      v16h bf[4];
#pragma unroll
      for (int ns = 0; ns < 4; ++ns) bf[ns] = frag_rm(lB + ns * 16 * LDT, LDT, lane);
      SCHED_FENCE();
#pragma unroll
      for (int ns = 0; ns < 4; ++ns) acc[ns] = WMMA16(a0, bf[ns], acc[ns]);
    }
    {
      const v16h a1 = frag_rm(lA + wid * 16 * LDT + 32, LDT, lane);
      v16h bf[4];
#pragma unroll
      for (int ns = 0; ns < 4; ++ns) bf[ns] = frag_rm(lB + ns * 16 * LDT + 32, LDT, lane);
      SCHED_FENCE();
#pragma unroll
      for (int ns = 0; ns < 4; ++ns) acc[ns] = WMMA16(a1, bf[ns], acc[ns]);
    }
  }
  const int hi = lane >> 4, nl = lane & 15;
#pragma unroll
  for (int ns = 0; ns < 4; ++ns) {
    const int n = n0 + ns * 16 + nl;
    const float bv = bias[n];
#pragma unroll
    for (int r = 0; r < 8; ++r) {
      const size_t m = (size_t)(m0 + wid * 16 + r + 8 * hi);
      const float y = fmaxf(acc[ns][r] + bv, 0.f) * (1.0f / 3.0f);
      out[m * D + n] = first ? y : (out[m * D + n] + y);
    }
  }
}

// ---------------------------------------------------------------------------
// Flash attention: grid (S/64, H, B), 4 waves/block, one 16-row q-tile/wave.
// K chunk row-major (QK^T B-frags contiguous); V chunk stored TRANSPOSED into
// LDS (PV B-frags contiguous). Online softmax with row stats replicated per
// 16-lane half group (shfl_xor reductions only).
// ---------------------------------------------------------------------------
__global__ __launch_bounds__(128)
void k_attn(const half_t* __restrict__ Q,
            const half_t* __restrict__ K,
            const half_t* __restrict__ V,
            half_t* __restrict__ O) {
  const int h = blockIdx.y, b = blockIdx.z;
  const int tid = threadIdx.x, wid = tid >> 5, lane = tid & 31;
  const int hi = lane >> 4, nl = lane & 15;
  const int q0 = blockIdx.x * 64 + wid * 16;
  const size_t base = (size_t)b * S * D + (size_t)h * DK;
  __shared__ __align__(16) half_t lK[32 * LDK];        // [krow][kd]
  __shared__ __align__(16) half_t lVt[64 * LDV];       // [dim][krow] (transposed)
  __shared__ __align__(16) half_t lP[4][16 * LDV];     // per-wave P tile [row][krow]

  // Q fragments (16x64 -> two 16x32 A-frags), straight from global
  v16h qa[2];
  {
    const int row = lane & 15, kb = (lane >> 4) << 3;
    const half_t* qp = Q + base + (size_t)(q0 + row) * D;
#pragma unroll
    for (int kk = 0; kk < 2; ++kk)
#pragma unroll
      for (int j = 0; j < 8; ++j) {
        qa[kk][j]     = qp[kk * 32 + kb + j];
        qa[kk][j + 8] = qp[kk * 32 + kb + 16 + j];
      }
  }

  const int cr = (tid * 16) >> 6, cc = (tid * 16) & 63;   // cc in {0,16,32,48}
  const half_t* gK = K + base + (size_t)cr * D + cc;
  const half_t* gV = V + base + (size_t)cr * D + cc;
  uint4* sk = (uint4*)(lK + cr * LDK + cc);

  __builtin_prefetch(gK, 0, 1);
  __builtin_prefetch(gV, 0, 1);

  v8f o[4] = {};
  float mrow[8], lrow[8];
#pragma unroll
  for (int r = 0; r < 8; ++r) { mrow[r] = -1e30f; lrow[r] = 0.f; }

  for (int kc = 0; kc < S; kc += 32) {
    __syncthreads();
    { // K chunk: vectorized; V chunk: transpose on store -> lVt[dim][krow]
      const uint4* p = (const uint4*)(gK + (size_t)kc * D);
      sk[0] = p[0]; sk[1] = p[1];
      const half_t* gv = gV + (size_t)kc * D;
#pragma unroll
      for (int j = 0; j < 16; ++j) lVt[(cc + j) * LDV + cr] = gv[j];
    }
    __syncthreads();
    if (kc + 32 < S) {
      __builtin_prefetch(gK + (size_t)(kc + 32) * D, 0, 1);
      __builtin_prefetch(gV + (size_t)(kc + 32) * D, 0, 1);
    }

    // scores: two 16x16 tiles (k-rows kc..kc+15 / kc+16..kc+31), K-dim = 64
    v8f sc[2] = {};
    {
      v16h b0 = frag_rm(lK + 0 * 16 * LDK,      LDK, lane);   // cb0, kk0
      v16h b1 = frag_rm(lK + 1 * 16 * LDK,      LDK, lane);   // cb1, kk0
      SCHED_FENCE();
      sc[0] = WMMA16(qa[0], b0, sc[0]);
      sc[1] = WMMA16(qa[0], b1, sc[1]);
    }
    {
      v16h b0 = frag_rm(lK + 0 * 16 * LDK + 32, LDK, lane);   // cb0, kk1
      v16h b1 = frag_rm(lK + 1 * 16 * LDK + 32, LDK, lane);   // cb1, kk1
      SCHED_FENCE();
      sc[0] = WMMA16(qa[1], b0, sc[0]);
      sc[1] = WMMA16(qa[1], b1, sc[1]);
    }

    // online softmax (rows r+8*hi; reductions within 16-lane half groups)
    float cm[8];
#pragma unroll
    for (int r = 0; r < 8; ++r) cm[r] = fmaxf(sc[0][r], sc[1][r]) * 0.125f;
#pragma unroll
    for (int off = 1; off < 16; off <<= 1)
#pragma unroll
      for (int r = 0; r < 8; ++r) cm[r] = fmaxf(cm[r], __shfl_xor(cm[r], off));

    float corr[8], psum[8];
#pragma unroll
    for (int r = 0; r < 8; ++r) {
      const float mn = fmaxf(mrow[r], cm[r]);
      corr[r] = __expf(mrow[r] - mn);
      mrow[r] = mn;
      const float p0 = __expf(sc[0][r] * 0.125f - mn);
      const float p1 = __expf(sc[1][r] * 0.125f - mn);
      lP[wid][(r + 8 * hi) * LDV + nl]      = (half_t)p0;
      lP[wid][(r + 8 * hi) * LDV + 16 + nl] = (half_t)p1;
      psum[r] = p0 + p1;
    }
#pragma unroll
    for (int off = 1; off < 16; off <<= 1)
#pragma unroll
      for (int r = 0; r < 8; ++r) psum[r] += __shfl_xor(psum[r], off);
#pragma unroll
    for (int r = 0; r < 8; ++r) lrow[r] = lrow[r] * corr[r] + psum[r];
#pragma unroll
    for (int ns = 0; ns < 4; ++ns)
#pragma unroll
      for (int r = 0; r < 8; ++r) o[ns][r] *= corr[r];

    __syncthreads();  // P visible (uniform across all 4 waves)

    // O += P(16x32) @ V(32x64), V^T tile gives contiguous B-frags
    const v16h pa = frag_rm(lP[wid], LDV, lane);
    {
      v16h v0 = frag_rm(lVt + 0 * 16 * LDV, LDV, lane);
      v16h v1 = frag_rm(lVt + 1 * 16 * LDV, LDV, lane);
      SCHED_FENCE();
      o[0] = WMMA16(pa, v0, o[0]);
      o[1] = WMMA16(pa, v1, o[1]);
    }
    {
      v16h v2 = frag_rm(lVt + 2 * 16 * LDV, LDV, lane);
      v16h v3 = frag_rm(lVt + 3 * 16 * LDV, LDV, lane);
      SCHED_FENCE();
      o[2] = WMMA16(pa, v2, o[2]);
      o[3] = WMMA16(pa, v3, o[3]);
    }
  }

  // normalize + store
#pragma unroll
  for (int ns = 0; ns < 4; ++ns)
#pragma unroll
    for (int r = 0; r < 8; ++r) {
      const float inv = 1.0f / lrow[r];
      O[base + (size_t)(q0 + r + 8 * hi) * D + ns * 16 + nl] = (half_t)(o[ns][r] * inv);
    }
}

// ---------------------------------------------------------------------------
// Small helper kernels
// ---------------------------------------------------------------------------
// out[n][k] = f16(w[k][n])  (transpose + convert, coalesced on write)
__global__ void k_w_to_f16t(const float* __restrict__ w, half_t* __restrict__ out) {
  const int i = blockIdx.x * 256 + threadIdx.x;
  if (i >= D * D) return;
  const int n = i / D, k = i - n * D;
  out[(size_t)n * D + k] = (half_t)w[(size_t)k * D + n];
}
__global__ void k_add(float* __restrict__ x, const float* __restrict__ y, int n) {
  const int i = blockIdx.x * 256 + threadIdx.x;
  if (i < n) x[i] += y[i];
}
__global__ void k_zero_pads(half_t* __restrict__ xpad) {
  const int i = blockIdx.x * 256 + threadIdx.x;           // B*4*D threads
  if (i >= B * 4 * D) return;
  const int bb = i / (4 * D);
  const int rem = i - bb * 4 * D;
  const int p = rem / D, dcol = rem - p * D;
  const int prow = (p < 2) ? p : (S + p);                 // rows 0,1,S+2,S+3
  xpad[((size_t)bb * SP + prow) * D + dcol] = (half_t)0.f;
}
// Wf[t][n][k] = conv_w[n][k][t] * g[n]/sqrt(v[n]+eps)   (transposed per tap)
__global__ void k_fold_conv(const float* __restrict__ w, const float* __restrict__ g,
                            const float* __restrict__ v, half_t* __restrict__ out,
                            int f, int total) {
  const int i = blockIdx.x * 256 + threadIdx.x;
  if (i >= total) return;
  const int k = i % D;
  const int rem = i / D;
  const int n = rem % D;
  const int t = rem / D;
  const float scl = g[n] * rsqrtf(v[n] + 1e-5f);
  out[((size_t)t * D + n) * D + k] = (half_t)(w[((size_t)n * D + k) * f + t] * scl);
}
// bias' = (conv_b - m)*g/sqrt(v+eps) + beta
__global__ void k_fold_bias(const float* __restrict__ cb, const float* __restrict__ g,
                            const float* __restrict__ bb, const float* __restrict__ m,
                            const float* __restrict__ v, float* __restrict__ out) {
  const int n = blockIdx.x * 256 + threadIdx.x;
  if (n >= D) return;
  out[n] = (cb[n] - m[n]) * g[n] * rsqrtf(v[n] + 1e-5f) + bb[n];
}

// ---------------------------------------------------------------------------
extern "C" void kernel_launch(void* const* d_in, const int* in_sizes, int n_in,
                              void* d_out, int out_size, void* d_ws, size_t ws_size,
                              hipStream_t stream) {
  (void)in_sizes; (void)n_in; (void)out_size; (void)ws_size;
  // setup_inputs() dict order:
  const float* in_x = (const float*)d_in[0];
  const float* wP[4] = {(const float*)d_in[1], (const float*)d_in[2],
                        (const float*)d_in[3], (const float*)d_in[4]}; // wq wk wv wo
  const float* bP[4] = {(const float*)d_in[5], (const float*)d_in[6],
                        (const float*)d_in[7], (const float*)d_in[8]}; // bq bk bv bo
  const int filt[3] = {1, 3, 5};
  const float *cw[3], *cb[3], *bg[3], *bbta[3], *bm[3], *bv[3];
  for (int i = 0; i < 3; ++i) {
    cw[i]   = (const float*)d_in[9 + 6 * i + 0];
    cb[i]   = (const float*)d_in[9 + 6 * i + 1];
    bg[i]   = (const float*)d_in[9 + 6 * i + 2];
    bbta[i] = (const float*)d_in[9 + 6 * i + 3];
    bm[i]   = (const float*)d_in[9 + 6 * i + 4];
    bv[i]   = (const float*)d_in[9 + 6 * i + 5];
  }
  const float* ln1a = (const float*)d_in[27];
  const float* ln2a = (const float*)d_in[28];
  const float* lnfa = (const float*)d_in[29];
  const float* ln1b = (const float*)d_in[30];
  const float* ln2b = (const float*)d_in[31];
  const float* lnfb = (const float*)d_in[32];

  // workspace carve-up
  char* wsb = (char*)d_ws;
  size_t off = 0;
  auto carve = [&](size_t bytes) { void* p = wsb + off; off = (off + bytes + 255) & ~(size_t)255; return p; };
  float*  x32    = (float*) carve((size_t)M_TOT * D * 4);
  half_t* hf     = (half_t*)carve((size_t)M_TOT * D * 2);
  half_t* qf     = (half_t*)carve((size_t)M_TOT * D * 2);
  half_t* kf     = (half_t*)carve((size_t)M_TOT * D * 2);
  half_t* vf     = (half_t*)carve((size_t)M_TOT * D * 2);
  half_t* attnf  = (half_t*)carve((size_t)M_TOT * D * 2);
  half_t* xpad   = (half_t*)carve((size_t)B * SP * D * 2);
  float*  cacc   = (float*) carve((size_t)M_TOT * D * 4);
  half_t* w16[4];
  for (int i = 0; i < 4; ++i) w16[i] = (half_t*)carve((size_t)D * D * 2);
  half_t* wc16[3];
  for (int i = 0; i < 3; ++i) wc16[i] = (half_t*)carve((size_t)filt[i] * D * D * 2);
  float* bf[3];
  for (int i = 0; i < 3; ++i) bf[i] = (float*)carve((size_t)D * 4);

  const int nElem = M_TOT * D;                   // 8 Mi
  const int eBlocks = (nElem + 255) / 256;

  // x working copy (inputs must not be mutated)
  hipMemcpyAsync(x32, in_x, (size_t)nElem * 4, hipMemcpyDeviceToDevice, stream);

  // weight preprocessing (deterministic, re-done every call)
  for (int i = 0; i < 4; ++i)
    k_w_to_f16t<<<(D * D + 255) / 256, 256, 0, stream>>>(wP[i], w16[i]);
  for (int i = 0; i < 3; ++i) {
    const int tot = filt[i] * D * D;
    k_fold_conv<<<(tot + 255) / 256, 256, 0, stream>>>(cw[i], bg[i], bv[i], wc16[i], filt[i], tot);
    k_fold_bias<<<(D + 255) / 256, 256, 0, stream>>>(cb[i], bg[i], bbta[i], bm[i], bv[i], bf[i]);
  }
  k_zero_pads<<<(B * 4 * D + 255) / 256, 256, 0, stream>>>(xpad);

  const dim3 gGemm(D / 64, M_TOT / 64);          // 16 x 128
  const dim3 gAttn(S / 64, H, B);                // 16 x 16 x 8

  for (int layer = 0; layer < 2; ++layer) {
    // --- MHA sublayer ---
    k_ln<<<M_TOT, 256, 0, stream>>>(x32, ln1a, ln1b, hf, nullptr, 0);
    k_gemm<<<gGemm, 128, 0, stream>>>(hf, w16[0], bP[0], qf, nullptr, 0);
    k_gemm<<<gGemm, 128, 0, stream>>>(hf, w16[1], bP[1], kf, nullptr, 0);
    k_gemm<<<gGemm, 128, 0, stream>>>(hf, w16[2], bP[2], vf, nullptr, 0);
    k_attn<<<gAttn, 128, 0, stream>>>(qf, kf, vf, attnf);
    k_gemm<<<gGemm, 128, 0, stream>>>(attnf, w16[3], bP[3], nullptr, x32, 1); // x += O-proj
    // --- conv-ensemble FFN sublayer ---
    k_ln<<<M_TOT, 256, 0, stream>>>(x32, ln2a, ln2b, xpad, nullptr, 1);
    for (int i = 0; i < 3; ++i)
      k_conv<<<gGemm, 128, 0, stream>>>(xpad, wc16[i], bf[i], cacc, filt[i], i == 0);
    k_add<<<eBlocks, 256, 0, stream>>>(x32, cacc, nElem);
  }
  // final LayerNorm -> f32 output
  k_ln<<<M_TOT, 256, 0, stream>>>(x32, lnfa, lnfb, nullptr, (float*)d_out, 2);
}
